// LSTM_20856361189689
// MI455X (gfx1250) — matrix operand order repaired and checked by reading, not compiled, not run on hardware
//
#include <hip/hip_runtime.h>
#include <hip/hip_bf16.h>

// ---------------------------------------------------------------------------
// LSTM for MI455X (gfx1250): bf16 WMMA (v_wmma_f32_16x16x32_bf16), f32 accum.
//  T=512, B=64, E=512, H=1024.  Gate order in z: [g, i, f, o].
//
//  Pipeline:
//   k0   convert embeds f32 -> bf16 (removes cvt VALU from GEMM hot loop)
//   k0t  transpose+convert Wx / Wh to bf16 [4H][K] (K-contiguous B operands)
//   k0i  zero h double-buffer + grid barrier words
//   k1   x_proj GEMM (bias folded in), output in WMMA C-register tiled layout
//   k2   persistent recurrent scan: Wh slice in LDS for all 512 steps,
//        cell state in registers, device-wide barrier per step.
// ---------------------------------------------------------------------------

#define T_DIM 512
#define B_DIM 64
#define E_DIM 512
#define H_DIM 1024
#define G4    (4 * H_DIM)      // 4096 fused gate columns
#define NWG   64               // workgroups in persistent recurrent kernel
#define LDSROW 2064            // bytes per (gate,col) LDS row: 1024*2 + 16 pad
                               // 2064 % 256 == 16 -> 16 lanes hit distinct banks

typedef __attribute__((ext_vector_type(16))) __bf16 v16bf;
typedef __attribute__((ext_vector_type(8)))  float  v8f;

union BfPack {
    v16bf v;
    uint4 q[2];
    unsigned short s[16];
};

union F8 {
    v8f    v;
    float4 f[2];
};

__device__ __forceinline__ unsigned short f2bf(float f) {
    unsigned u = __float_as_uint(f);
    unsigned r = 0x7FFFu + ((u >> 16) & 1u);   // round-to-nearest-even
    return (unsigned short)((u + r) >> 16);
}

// ---------------------------------------------------------------------------
// Kernel 0: convert embeds f32 -> bf16 (same [row][k] layout, 4 elems/thread)
// ---------------------------------------------------------------------------
__global__ void cvt_embeds(const float* __restrict__ in, unsigned short* __restrict__ out) {
    size_t id = ((size_t)blockIdx.x * blockDim.x + threadIdx.x) * 4;
    float4 f = *(const float4*)(in + id);
    union { uint2 u; unsigned short s[4]; } p;
    p.s[0] = f2bf(f.x); p.s[1] = f2bf(f.y); p.s[2] = f2bf(f.z); p.s[3] = f2bf(f.w);
    *(uint2*)(out + id) = p.u;
}

// ---------------------------------------------------------------------------
// Kernel 0t: transpose+convert 4 gate weight matrices [K][C] f32 -> bf16 [4][C][K]
// ---------------------------------------------------------------------------
__global__ void transpose_bf16(const float* __restrict__ W0, const float* __restrict__ W1,
                               const float* __restrict__ W2, const float* __restrict__ W3,
                               unsigned short* __restrict__ out, int K, int C) {
    size_t id = (size_t)blockIdx.x * blockDim.x + threadIdx.x;   // k fastest
    int k = (int)(id % (size_t)K);
    int c = (int)((id / (size_t)K) % (size_t)C);
    int g = (int)(id / ((size_t)K * C));
    const float* W = (g == 0) ? W0 : (g == 1) ? W1 : (g == 2) ? W2 : W3;
    out[id] = f2bf(W[(size_t)k * C + c]);
}

// ---------------------------------------------------------------------------
// Kernel 0i: zero h double-buffer (bf16) and grid-barrier words
// ---------------------------------------------------------------------------
__global__ void init_ws(unsigned short* __restrict__ hbuf, unsigned int* __restrict__ bar) {
    int id = blockIdx.x * blockDim.x + threadIdx.x;
    if (id < 2 * B_DIM * H_DIM) hbuf[id] = 0;
    if (id < 2) bar[id] = 0;
}

// ---------------------------------------------------------------------------
// Kernel 1: x_proj = embeds @ Wx + bias   (bf16 WMMA, f32 acc)
// One wave per 16x16 output tile x 4 gates (A chunk reused across 4 WMMAs).
// Output layout = WMMA C-register tiling: [waveTile][gate][lane][8] f32 so the
// recurrent kernel reads its accumulator init as one contiguous 32B per lane.
//   waveTile = mTile*64 + nSub;  mTile = t*4 + bTile (rows of T*B), nSub = H/16.
// ---------------------------------------------------------------------------
__global__ void xproj_kernel(const unsigned short* __restrict__ ebf,   // [T*B][E] bf16
                             const unsigned short* __restrict__ Wxt,   // [4H][E] bf16
                             const float* __restrict__ b_g, const float* __restrict__ b_i,
                             const float* __restrict__ b_f, const float* __restrict__ b_o,
                             float* __restrict__ xproj) {
    const int lane = threadIdx.x & 31;
    const int wave = threadIdx.x >> 5;
    const int waveTile = blockIdx.x * 8 + wave;
    const int mTile = waveTile >> 6;            // 0..2047 : 16-row tile of (T*B)
    const int nSub  = waveTile & 63;            // 0..63   : 16-col group within H
    const int half  = lane >> 4;
    const int lm    = lane & 15;
    const int rowA  = mTile * 16 + lm;          // A row this lane supplies
    const int ncol  = nSub * 16 + lm;           // within-gate column (N = lane%16)

    v8f acc[4];
    {
        const float bg = b_g[ncol], bi = b_i[ncol], bf = b_f[ncol], bo = b_o[ncol];
        #pragma unroll
        for (int r = 0; r < 8; ++r) { acc[0][r] = bg; acc[1][r] = bi; acc[2][r] = bf; acc[3][r] = bo; }
    }

    const unsigned short* arow = ebf + (size_t)rowA * E_DIM;
    #pragma unroll 2
    for (int kk = 0; kk < E_DIM / 32; ++kk) {
        // A: half 0 -> K {0..7,16..23}, half 1 -> {8..15,24..31} (ISA interleave)
        BfPack a, bb[4];
        const unsigned short* ap = arow + kk * 32 + half * 8;
        a.q[0] = *(const uint4*)(ap);
        a.q[1] = *(const uint4*)(ap + 16);
        #pragma unroll
        for (int g = 0; g < 4; ++g) {           // batch all B loads (one clause)
            const unsigned short* bp =
                Wxt + ((size_t)(g * H_DIM + ncol)) * E_DIM + kk * 32 + half * 8;
            bb[g].q[0] = *(const uint4*)(bp);
            bb[g].q[1] = *(const uint4*)(bp + 16);
        }
        #pragma unroll
        for (int g = 0; g < 4; ++g)             // then WMMAs back-to-back
            acc[g] = __builtin_amdgcn_wmma_f32_16x16x32_bf16(
                false, a.v, false, bb[g].v, (short)0, acc[g], false, false);
    }

    // coalesced tiled store: [waveTile][gate][lane][8]
    #pragma unroll
    for (int g = 0; g < 4; ++g) {
        float* dst = xproj + ((size_t)waveTile * 4 + g) * 256 + lane * 8;
        F8 u; u.v = acc[g];
        *(float4*)(dst)     = u.f[0];
        *(float4*)(dst + 4) = u.f[1];
    }
}

// ---------------------------------------------------------------------------
// Kernel 2: persistent recurrent scan.
//  64 WGs x 128 threads (4 waves).  WG s owns H columns [16s, 16s+16).
//  Wh slice for all 4 gates lives in LDS (bf16, 129 KB) for all 512 steps.
//  Each wave: 16 batch rows x 16 H cols, 4 gate accumulators, c in registers.
//  Device-wide sync per step via generation-counter barrier in d_ws.
// ---------------------------------------------------------------------------
__global__ void lstm_recurrent(const float* __restrict__ xproj,   // tiled layout
                               const unsigned short* __restrict__ Wht,   // [4H][H] bf16
                               unsigned short* __restrict__ hbuf,        // [2][B][H] bf16
                               float* __restrict__ h_arr,                // [T][B][H]
                               float* __restrict__ h_last,               // [B][H]
                               unsigned int* __restrict__ bar) {
    extern __shared__ unsigned char smem[];
    const int s    = blockIdx.x;        // H-slice id 0..63
    const int tid  = threadIdx.x;       // 0..127
    const int lane = tid & 31;
    const int wave = tid >> 5;          // 0..3
    const int half = lane >> 4;
    const int lm   = lane & 15;

    // ---- one-time: load Wh slice into LDS as [gate][n][k] bf16, padded rows ----
    for (int i = tid; i < 8192; i += 128) {          // 8192 x uint4 (8 bf16 each)
        int g  = i >> 11;                            // 2048 chunks per gate
        int n  = (i >> 7) & 15;                      // 128 chunks per column
        int kc = i & 127;
        uint4 src = *(const uint4*)(Wht + ((size_t)(g * H_DIM + s * 16 + n)) * H_DIM + kc * 8);
        *(uint4*)(smem + (g * 16 + n) * LDSROW + kc * 16) = src;
    }
    __syncthreads();

    const int Mbase = wave * 16;
    const int rowA  = Mbase + lm;          // batch row this lane feeds into A
    const int crow  = Mbase + half * 8;    // C-layout base batch row
    const int ncol  = s * 16 + lm;         // global H column for this lane's C

    v8f cst = {};                          // cell state tile, f32, lives 512 steps

    for (int t = 0; t < T_DIM; ++t) {
        const unsigned short* hin  = hbuf + (size_t)(t & 1) * (B_DIM * H_DIM);
        unsigned short*       hout = hbuf + (size_t)((t + 1) & 1) * (B_DIM * H_DIM);

        // acc init = x_proj (bias folded in); tiled layout -> 32B/lane/gate
        const size_t tIdx = ((size_t)(t * 4 + wave) * 64 + s);
        v8f acc[4];
        #pragma unroll
        for (int g = 0; g < 4; ++g) {
            const float* p = xproj + (tIdx * 4 + g) * 256 + lane * 8;
            F8 u;
            u.f[0] = *(const float4*)(p);
            u.f[1] = *(const float4*)(p + 4);
            acc[g] = u.v;
        }

        // prefetch next step's x_proj lines (HBM stream) -> global_prefetch_b8
        if (t + 1 < T_DIM) {
            const float* pn = xproj + (((size_t)((t + 1) * 4 + wave) * 64 + s) * 4) * 256 + lane * 8;
            #pragma unroll
            for (int g = 0; g < 4; ++g)
                __builtin_prefetch(pn + g * 256, 0, 1);
        }

        const unsigned short* hrow = hin + (size_t)rowA * H_DIM;
        #pragma unroll 4
        for (int kk = 0; kk < H_DIM / 32; ++kk) {
            BfPack a, bb[4];
            const unsigned short* ap = hrow + kk * 32 + half * 8;
            a.q[0] = *(const uint4*)(ap);            // A from h(t): 2 x b128
            a.q[1] = *(const uint4*)(ap + 16);
            const unsigned char* lb = smem + kk * 64 + half * 16;
            #pragma unroll
            for (int g = 0; g < 4; ++g) {            // batch LDS B loads
                const unsigned char* bp = lb + (g * 16 + lm) * LDSROW;
                bb[g].q[0] = *(const uint4*)(bp);
                bb[g].q[1] = *(const uint4*)(bp + 32);
            }
            #pragma unroll
            for (int g = 0; g < 4; ++g)              // WMMAs back-to-back
                acc[g] = __builtin_amdgcn_wmma_f32_16x16x32_bf16(
                    false, a.v, false, bb[g].v, (short)0, acc[g], false, false);
        }

        // gates + state update, all in-lane (C layouts of the 4 accs coincide)
        float* outp = h_arr + (size_t)t * B_DIM * H_DIM;
        #pragma unroll
        for (int r = 0; r < 8; ++r) {
            float gg = tanhf(acc[0][r]);
            float ii = 1.f / (1.f + __expf(-acc[1][r]));
            float ff = 1.f / (1.f + __expf(-acc[2][r]));
            float oo = 1.f / (1.f + __expf(-acc[3][r]));
            float c  = gg * ii + cst[r] * ff;
            cst[r]   = c;
            float h  = tanhf(c) * oo;
            size_t o = (size_t)(crow + r) * H_DIM + ncol;
            outp[o]  = h;
            hout[o]  = f2bf(h);
            if (t == T_DIM - 1) h_last[o] = h;
        }
        if (t == T_DIM - 1) break;

        // ---- device-wide barrier (generation counter = completed steps) ----
        __threadfence();
        __syncthreads();
        if (tid == 0) {
            unsigned arrived = __hip_atomic_fetch_add(&bar[0], 1u, __ATOMIC_ACQ_REL,
                                                      __HIP_MEMORY_SCOPE_AGENT);
            if (arrived == NWG - 1u) {
                __hip_atomic_store(&bar[0], 0u, __ATOMIC_RELAXED, __HIP_MEMORY_SCOPE_AGENT);
                __hip_atomic_fetch_add(&bar[1], 1u, __ATOMIC_RELEASE, __HIP_MEMORY_SCOPE_AGENT);
            } else {
                while (__hip_atomic_load(&bar[1], __ATOMIC_ACQUIRE,
                                         __HIP_MEMORY_SCOPE_AGENT) < (unsigned)(t + 1)) {
                    __builtin_amdgcn_s_sleep(1);
                }
            }
        }
        __syncthreads();
    }
}

// ---------------------------------------------------------------------------
extern "C" void kernel_launch(void* const* d_in, const int* in_sizes, int n_in,
                              void* d_out, int out_size, void* d_ws, size_t ws_size,
                              hipStream_t stream) {
    const float* embeds = (const float*)d_in[0];
    const float* W_fx = (const float*)d_in[1];
    const float* W_fh = (const float*)d_in[2];
    const float* b_f  = (const float*)d_in[3];
    const float* W_ix = (const float*)d_in[4];
    const float* W_ih = (const float*)d_in[5];
    const float* b_i  = (const float*)d_in[6];
    const float* W_gx = (const float*)d_in[7];
    const float* W_gh = (const float*)d_in[8];
    const float* b_g  = (const float*)d_in[9];
    const float* W_ox = (const float*)d_in[10];
    const float* W_oh = (const float*)d_in[11];
    const float* b_o  = (const float*)d_in[12];

    unsigned char* ws = (unsigned char*)d_ws;
    float* xproj = (float*)ws;                                        // 512 MB tiled
    size_t off = (size_t)T_DIM * B_DIM * G4 * sizeof(float);
    unsigned short* ebf  = (unsigned short*)(ws + off); off += (size_t)T_DIM * B_DIM * E_DIM * 2;
    unsigned short* Wxt  = (unsigned short*)(ws + off); off += (size_t)4 * H_DIM * E_DIM * 2;
    unsigned short* Wht  = (unsigned short*)(ws + off); off += (size_t)4 * H_DIM * H_DIM * 2;
    unsigned short* hbuf = (unsigned short*)(ws + off); off += (size_t)2 * B_DIM * H_DIM * 2;
    unsigned int*   bar  = (unsigned int*)(ws + off);

    float* h_arr  = (float*)d_out;
    float* h_last = h_arr + (size_t)T_DIM * B_DIM * H_DIM;

    cvt_embeds<<<(T_DIM * B_DIM * E_DIM / 4) / 256, 256, 0, stream>>>(embeds, ebf);
    // gate order [g, i, f, o]
    transpose_bf16<<<(4 * H_DIM * E_DIM) / 256, 256, 0, stream>>>(
        W_gx, W_ix, W_fx, W_ox, Wxt, E_DIM, H_DIM);
    transpose_bf16<<<(4 * H_DIM * H_DIM) / 256, 256, 0, stream>>>(
        W_gh, W_ih, W_fh, W_oh, Wht, H_DIM, H_DIM);
    init_ws<<<(2 * B_DIM * H_DIM) / 256, 256, 0, stream>>>(hbuf, bar);

    xproj_kernel<<<16384, 256, 0, stream>>>(ebf, Wxt, b_g, b_i, b_f, b_o, xproj);

    const int ldsBytes = 4 * 16 * LDSROW;   // 132096 B (< 320 KB WGP LDS)
    hipFuncSetAttribute(reinterpret_cast<const void*>(lstm_recurrent),
                        hipFuncAttributeMaxDynamicSharedMemorySize, ldsBytes);
    lstm_recurrent<<<NWG, 128, ldsBytes, stream>>>(xproj, Wht, hbuf, h_arr, h_last, bar);
}
